// CombinedGraphLayer_33724083208430
// MI455X (gfx1250) — compile-verified, matrix operand order checked by reading, not run on hardware
//
#include <hip/hip_runtime.h>
#include <stdint.h>

typedef __bf16 bf16;
typedef __attribute__((ext_vector_type(16))) __bf16   v16bf;
typedef __attribute__((ext_vector_type(8)))  float    v8f;
typedef __attribute__((ext_vector_type(8)))  uint32_t v8u;

constexpr int cB    = 8;
constexpr int cNp   = 25600;     // N (already multiple of 128 -> no pad)
constexpr int cD    = 256;
constexpr int cH    = 128;
constexpr int cDD   = 128;
constexpr int cNB   = 200;       // bins
constexpr int cBIN  = 128;       // bin size
constexpr int cBN   = cB * cNp;  // 204800 rows
constexpr int cNKEY = 400;       // LSH keys 0..398 (argmax + (msk?0:199))
constexpr float cEPS = 1e-6f;
constexpr float cDMUL = 0.1f;

__device__ __forceinline__ float eluf(float x)     { return x > 0.f ? x : (__expf(x) - 1.f); }
__device__ __forceinline__ float sigmoidf(float x) { return 1.f / (1.f + __expf(-x)); }

// ---------------------------------------------------------------------------
// CDNA5 WMMA bf16 fragment loader. A-operand layout for 16x32 (MxK) bf16:
//   lanes 0-15 : M = lane,    K in {0..7, 16..23}  (2 per VGPR)
//   lanes 16-31: M = lane-16, K in {8..15, 24..31}
// B-operand (32x16 KxN) uses the identical pattern with "row" = N column, so
// we always load B from a pre-transposed [N,K] row-major buffer -> every load
// vectorizes to two global_load_b128.
// ---------------------------------------------------------------------------
__device__ __forceinline__ v16bf ld_frag(const bf16* __restrict__ base, int ld,
                                         int row, int k0) {
  const int lane = threadIdx.x & 31;
  const int half = lane >> 4;
  const bf16* p = base + (size_t)(row + (lane & 15)) * ld + k0 + 8 * half;
  v8u u;
  u[0] = *(const uint32_t*)(p + 0);
  u[1] = *(const uint32_t*)(p + 2);
  u[2] = *(const uint32_t*)(p + 4);
  u[3] = *(const uint32_t*)(p + 6);
  u[4] = *(const uint32_t*)(p + 16);
  u[5] = *(const uint32_t*)(p + 18);
  u[6] = *(const uint32_t*)(p + 20);
  u[7] = *(const uint32_t*)(p + 22);
  return __builtin_bit_cast(v16bf, u);
}

__device__ __forceinline__ v8f wmma_one(v16bf a, v16bf b, v8f c) {
  return __builtin_amdgcn_wmma_f32_16x16x32_bf16(false, a, false, b,
                                                 (short)0, c, false, false);
}

// 2x2 register-blocked 32x32 macro-tile: 4 WMMAs per 4 fragment loads.
__device__ __forceinline__ void wmma_2x2(const bf16* __restrict__ A, int lda, int mT,
                                         const bf16* __restrict__ BT, int ldbt, int nT,
                                         int K, v8f& a00, v8f& a01, v8f& a10, v8f& a11) {
  for (int k = 0; k < K; k += 32) {
    v16bf A0 = ld_frag(A, lda, mT, k);
    v16bf A1 = ld_frag(A, lda, mT + 16, k);
    v16bf B0 = ld_frag(BT, ldbt, nT, k);
    v16bf B1 = ld_frag(BT, ldbt, nT + 16, k);
    a00 = wmma_one(A0, B0, a00);
    a01 = wmma_one(A0, B1, a01);
    a10 = wmma_one(A1, B0, a10);
    a11 = wmma_one(A1, B1, a11);
  }
}
// D layout: VGPR r -> lanes 0-15: (M=r, N=lane); lanes 16-31: (M=r+8, N=lane-16)

// ---------------------------------------------------------------------------
__global__ void k_transpose_bf16(const float* __restrict__ src, bf16* __restrict__ dst,
                                 int K, int N) {
  int idx = blockIdx.x * 256 + threadIdx.x;
  if (idx >= K * N) return;
  int k = idx / N, n = idx % N;
  dst[(size_t)n * K + k] = (bf16)src[idx];
}

__global__ void k_layernorm(const float* __restrict__ x, const float* __restrict__ g,
                            const float* __restrict__ beta, float* __restrict__ xnF,
                            bf16* __restrict__ xnB) {
  const int row = blockIdx.x;
  const int t = threadIdx.x;            // 256 threads, one element each
  float v = x[(size_t)row * cD + t];
  __shared__ float red[8], red2[8];
  float s = v;
  for (int o = 16; o > 0; o >>= 1) s += __shfl_xor(s, o);
  if ((t & 31) == 0) red[t >> 5] = s;
  __syncthreads();
  float mu = 0.f;
  for (int w = 0; w < 8; w++) mu += red[w];
  mu *= (1.f / cD);
  float d = v - mu;
  float s2 = d * d;
  for (int o = 16; o > 0; o >>= 1) s2 += __shfl_xor(s2, o);
  if ((t & 31) == 0) red2[t >> 5] = s2;
  __syncthreads();
  float var = 0.f;
  for (int w = 0; w < 8; w++) var += red2[w];
  var *= (1.f / cD);
  float xn = d * rsqrtf(var + cEPS) * g[t] + beta[t];
  xnF[(size_t)row * cD + t] = xn;
  xnB[(size_t)row * cD + t] = (bf16)xn;
}

// FFN GEMM: out = ELU(A[M,K] @ W + bias), W given transposed [N,K] bf16.
// One wave computes a 32x32 macro-tile.
__global__ void k_gemm_ffn(const bf16* __restrict__ A, const bf16* __restrict__ WT,
                           const float* __restrict__ bias, bf16* __restrict__ outB,
                           float* __restrict__ outF, int M, int N, int K) {
  const int wid = threadIdx.x >> 5;
  const int lane = threadIdx.x & 31;
  const int half = lane >> 4;
  const int tilesN = N >> 5;
  int tile = blockIdx.x * 8 + wid;
  int mT = (tile / tilesN) << 5;
  int nT = (tile % tilesN) << 5;
  v8f a00 = {}, a01 = {}, a10 = {}, a11 = {};
  wmma_2x2(A, K, mT, WT, K, nT, K, a00, a01, a10, a11);
  int nl = lane & 15;
  for (int j = 0; j < 2; j++) {
    int n = nT + 16 * j + nl;
    float bn = bias[n];
    v8f t0 = j ? a01 : a00;
    v8f t1 = j ? a11 : a10;
    for (int r = 0; r < 8; r++) {
      int m0 = mT + r + 8 * half;
      int m1 = m0 + 16;
      float v0 = eluf(t0[r] + bn);
      float v1 = eluf(t1[r] + bn);
      if (outB) { outB[(size_t)m0 * N + n] = (bf16)v0; outB[(size_t)m1 * N + n] = (bf16)v1; }
      if (outF) { outF[(size_t)m0 * N + n] = v0;       outF[(size_t)m1 * N + n] = v1; }
    }
  }
}

// 256x256 graph-conv GEMMs over all rows.  mode 0: (xmk@th)*mskf -> yT (bf16,
// stored transposed per bin [256][128]); mode 1: xmk@wh -> f32; mode 2:
// sigmoid(xb@wt + bt) -> f32.  One wave computes a 32x32 macro-tile.
__global__ void k_gemm_big(const bf16* __restrict__ A, const bf16* __restrict__ WT,
                           const float* __restrict__ bias, const float* __restrict__ mskf,
                           bf16* __restrict__ outYT, float* __restrict__ outF, int mode) {
  const int wid = threadIdx.x >> 5;
  const int lane = threadIdx.x & 31;
  const int half = lane >> 4;
  const int tilesN = cD >> 5;   // 8
  int tile = blockIdx.x * 8 + wid;
  int mT = (tile / tilesN) << 5;
  int nT = (tile % tilesN) << 5;
  v8f a00 = {}, a01 = {}, a10 = {}, a11 = {};
  wmma_2x2(A, cD, mT, WT, cD, nT, cD, a00, a01, a10, a11);
  int nl = lane & 15;
  for (int j = 0; j < 2; j++) {
    int n = nT + 16 * j + nl;
    v8f t0 = j ? a01 : a00;
    v8f t1 = j ? a11 : a10;
    for (int r = 0; r < 8; r++) {
      int m0 = mT + r + 8 * half;
      int m1 = m0 + 16;
      float v0 = t0[r], v1 = t1[r];
      if (mode == 0) {
        v0 *= mskf[m0];
        v1 *= mskf[m1];
        outYT[(size_t)(m0 >> 7) * (cD * cBIN) + (size_t)n * cBIN + (m0 & 127)] = (bf16)v0;
        outYT[(size_t)(m1 >> 7) * (cD * cBIN) + (size_t)n * cBIN + (m1 & 127)] = (bf16)v1;
      } else if (mode == 1) {
        outF[(size_t)m0 * cD + n] = v0;
        outF[(size_t)m1 * cD + n] = v1;
      } else {
        float bn = bias[n];
        outF[(size_t)m0 * cD + n] = sigmoidf(v0 + bn);
        outF[(size_t)m1 * cD + n] = sigmoidf(v1 + bn);
      }
    }
  }
}

// LSH: argmax over [x_dist@cb, -x_dist@cb] (200 keys), +199 if masked-out.
__global__ void k_lsh(const float* __restrict__ xd, const float* __restrict__ cb,
                      const int* __restrict__ msk, int* __restrict__ binIdx) {
  __shared__ float rb[8][cDD];
  const int wid = threadIdx.x >> 5;
  const int lane = threadIdx.x & 31;
  const int row = blockIdx.x * 8 + wid;
  for (int d = lane; d < cDD; d += 32) rb[wid][d] = xd[(size_t)row * cDD + d];
  __syncthreads();
  float best = -3.4e38f;
  int bidx = 0x7fffffff;
  for (int j = lane; j < 100; j += 32) {
    float s = 0.f;
    for (int d = 0; d < cDD; d++) s += rb[wid][d] * cb[d * 100 + j];
    if (s > best || (s == best && j < bidx)) { best = s; bidx = j; }
    float ns = -s;
    int j2 = j + 100;
    if (ns > best || (ns == best && j2 < bidx)) { best = ns; bidx = j2; }
  }
  for (int o = 16; o > 0; o >>= 1) {
    float ov = __shfl_xor(best, o);
    int   oi = __shfl_xor(bidx, o);
    if (ov > best || (ov == best && oi < bidx)) { best = ov; bidx = oi; }
  }
  if (lane == 0) binIdx[row] = bidx + (msk[row] != 0 ? 0 : (cNB - 1));
}

__global__ void k_zero(int* p, int n) {
  int i = blockIdx.x * 256 + threadIdx.x;
  if (i < n) p[i] = 0;
}
__global__ void k_hist(const int* __restrict__ binIdx, int* __restrict__ counts) {
  int i = blockIdx.x * 256 + threadIdx.x;
  if (i >= cBN) return;
  atomicAdd(&counts[(i / cNp) * cNKEY + binIdx[i]], 1);
}
__global__ void k_scan(const int* __restrict__ counts, int* __restrict__ offs) {
  int b = threadIdx.x;
  if (b >= cB) return;
  int acc = 0;
  for (int k = 0; k < cNKEY; k++) { offs[b * cNKEY + k] = acc; acc += counts[b * cNKEY + k]; }
}
// Deterministic stable counting-sort rank: one wave per (batch,key), ballot prefix.
__global__ void k_rank(const int* __restrict__ binIdx, const int* __restrict__ offs,
                       int* __restrict__ split) {
  int b = blockIdx.x / cNKEY;
  int key = blockIdx.x % cNKEY;
  int lane = threadIdx.x;
  int base = offs[b * cNKEY + key];
  int running = 0;
  for (int c = 0; c < cNp; c += 32) {
    int i = c + lane;
    bool match = (binIdx[b * cNp + i] == key);
    unsigned long long m = __ballot(match);
    if (match) {
      int prefix = __popcll(m & ((1ull << lane) - 1ull));
      split[b * cNp + base + running + prefix] = i;
    }
    running += __popcll(m);
  }
}

__global__ void k_gather(const int* __restrict__ split, const int* __restrict__ msk,
                         const float* __restrict__ xnF, const float* __restrict__ xd,
                         float* __restrict__ mskf, float* __restrict__ nsq,
                         bf16* __restrict__ xm, float* __restrict__ xbF,
                         bf16* __restrict__ xbB, bf16* __restrict__ xbmB) {
  const int row = blockIdx.x;         // binned slot (global)
  const int t = threadIdx.x;          // 256
  const int b = row / cNp;
  const int i = split[row];
  const size_t src = (size_t)b * cNp + i;
  const float m = (msk[src] != 0) ? 1.f : 0.f;
  float xv = xnF[src * cD + t];
  xbF [(size_t)row * cD + t] = xv;
  xbB [(size_t)row * cD + t] = (bf16)xv;
  xbmB[(size_t)row * cD + t] = (bf16)(xv * m);
  __shared__ float part[256];
  float dv = 0.f;
  if (t < cDD) {
    dv = xd[src * cDD + t] * m;
    xm[(size_t)row * cDD + t] = (bf16)dv;
  }
  part[t] = dv * dv;
  __syncthreads();
  for (int s = 128; s > 0; s >>= 1) {
    if (t < s) part[t] += part[t + s];
    __syncthreads();
  }
  if (t == 0) { nsq[row] = part[0]; mskf[row] = m; }
}

// Per-bin Gaussian kernel: inner = xk@xk^T via WMMA (both operands are rows of
// xk), 32x32 macro-tiles, fused sqrt/exp/clamp/mask epilogue.
__global__ void k_dm(const bf16* __restrict__ xm, const float* __restrict__ nsq,
                     const float* __restrict__ mskf, bf16* __restrict__ dm) {
  const int binG = blockIdx.x;
  const int wid = threadIdx.x >> 5;
  const int lane = threadIdx.x & 31;
  const int half = lane >> 4;
  const size_t base = (size_t)binG * cBIN;
  const bf16* xk = xm + base * cDD;
  for (int t = wid; t < 16; t += 8) {           // 4x4 macro-tiles of 32x32
    int pT = (t >> 2) << 5;
    int qT = (t & 3) << 5;
    v8f a00 = {}, a01 = {}, a10 = {}, a11 = {};
    wmma_2x2(xk, cDD, pT, xk, cDD, qT, cDD, a00, a01, a10, a11);
    int nl = lane & 15;
    for (int j = 0; j < 2; j++) {
      int q = qT + 16 * j + nl;
      float nq = nsq[base + q];
      float mq = mskf[base + q];
      v8f t0 = j ? a01 : a00;
      v8f t1 = j ? a11 : a10;
      for (int r = 0; r < 8; r++) {
        int p0 = pT + r + 8 * half;
        int p1 = p0 + 16;
        float d2a = nsq[base + p0] - 2.f * t0[r] + nq;
        float d2b = nsq[base + p1] - 2.f * t1[r] + nq;
        d2a = fminf(fmaxf(d2a, 1e-6f), 1e6f);
        d2b = fminf(fmaxf(d2b, 1e-6f), 1e6f);
        float va = fminf(fmaxf(__expf(-cDMUL * sqrtf(d2a)), 0.f), 1.f) * mskf[base + p0] * mq;
        float vb = fminf(fmaxf(__expf(-cDMUL * sqrtf(d2b)), 0.f), 1.f) * mskf[base + p1] * mq;
        dm[(size_t)binG * (cBIN * cBIN) + (size_t)p0 * cBIN + q] = (bf16)va;
        dm[(size_t)binG * (cBIN * cBIN) + (size_t)p1 * cBIN + q] = (bf16)vb;
      }
    }
  }
}

// Per-bin f_hom = dm @ y (y supplied transposed per bin) + full gated combine.
__global__ void k_hom(const bf16* __restrict__ dm, const bf16* __restrict__ yT,
                      const float* __restrict__ gate, const float* __restrict__ fhet,
                      const float* __restrict__ mskf, float* __restrict__ xbF,
                      bf16* __restrict__ xbB, bf16* __restrict__ xbmB) {
  const int binG = blockIdx.x;
  const int wid = threadIdx.x >> 5;
  const int lane = threadIdx.x & 31;
  const int half = lane >> 4;
  const bf16* Adm = dm + (size_t)binG * (cBIN * cBIN);
  const bf16* Byt = yT + (size_t)binG * (cD * cBIN);
  for (int t = wid; t < 32; t += 8) {           // 4x8 macro-tiles of 32x32
    int pT = (t >> 3) << 5;
    int nT = (t & 7) << 5;
    v8f a00 = {}, a01 = {}, a10 = {}, a11 = {};
    wmma_2x2(Adm, cBIN, pT, Byt, cBIN, nT, cBIN, a00, a01, a10, a11);
    int nl = lane & 15;
    for (int j = 0; j < 2; j++) {
      int n = nT + 16 * j + nl;
      v8f t0 = j ? a01 : a00;
      v8f t1 = j ? a11 : a10;
      for (int r = 0; r < 8; r++) {
        int p0 = pT + r + 8 * half;
        int p1 = p0 + 16;
        size_t row0 = (size_t)binG * cBIN + p0;
        size_t row1 = (size_t)binG * cBIN + p1;
        float g0 = gate[row0 * cD + n], fh0 = fhet[row0 * cD + n];
        float g1 = gate[row1 * cD + n], fh1 = fhet[row1 * cD + n];
        float v0 = eluf(g0 * t0[r] + (1.f - g0) * fh0) * mskf[row0];
        float v1 = eluf(g1 * t1[r] + (1.f - g1) * fh1) * mskf[row1];
        xbF[row0 * cD + n] = v0;
        xbF[row1 * cD + n] = v1;
        bf16 vb0 = (bf16)v0, vb1 = (bf16)v1;
        xbB [row0 * cD + n] = vb0;  xbB [row1 * cD + n] = vb1;
        xbmB[row0 * cD + n] = vb0;  xbmB[row1 * cD + n] = vb1;  // already masked
      }
    }
  }
}

__global__ void k_scatter(const int* __restrict__ split, const float* __restrict__ xbF,
                          float* __restrict__ out) {
  const int row = blockIdx.x;
  const int t = threadIdx.x;
  const int b = row / cNp;
  const int i = split[row];
  out[((size_t)b * cNp + i) * cD + t] = xbF[(size_t)row * cD + t];
}

// ---------------------------------------------------------------------------
extern "C" void kernel_launch(void* const* d_in, const int* in_sizes, int n_in,
                              void* d_out, int out_size, void* d_ws, size_t ws_size,
                              hipStream_t stream) {
  (void)in_sizes; (void)n_in; (void)out_size; (void)ws_size;
  const float* x    = (const float*)d_in[0];
  const int*   msk  = (const int*)  d_in[1];
  const float* ln_g = (const float*)d_in[2];
  const float* ln_b = (const float*)d_in[3];
  const float* w1   = (const float*)d_in[4];
  const float* b1   = (const float*)d_in[5];
  const float* w2   = (const float*)d_in[6];
  const float* b2   = (const float*)d_in[7];
  const float* w3   = (const float*)d_in[8];
  const float* b3   = (const float*)d_in[9];
  const float* cb   = (const float*)d_in[10];
  const float* thW[2] = { (const float*)d_in[11], (const float*)d_in[15] };
  const float* whW[2] = { (const float*)d_in[12], (const float*)d_in[16] };
  const float* wtW[2] = { (const float*)d_in[13], (const float*)d_in[17] };
  const float* btW[2] = { (const float*)d_in[14], (const float*)d_in[18] };
  float* out = (float*)d_out;

  char* w = (char*)d_ws;
  size_t o = 0;
  auto take = [&](size_t bytes) -> char* {
    char* p = w + o; o += (bytes + 255) & ~(size_t)255; return p;
  };
  const size_t SZ_F32_D = (size_t)cBN * cD * 4;   // 209.7 MB
  const size_t SZ_BF_D  = (size_t)cBN * cD * 2;   // 104.9 MB
  const size_t SZ_F32_H = (size_t)cBN * cH * 4;   // 104.9 MB
  const size_t SZ_BF_H  = (size_t)cBN * cH * 2;   //  52.4 MB

  // region 1: xn f32 (phase 1)  ==aliased==  f_het f32 (phase 2)
  char* r1 = take(SZ_F32_D);
  float* xnF  = (float*)r1;
  float* fhet = (float*)r1;
  // region 2: xn bf16 + h1 + h2 (phase 1)  ==aliased==  gate f32 (phase 2)
  char* r2 = take(SZ_F32_D);
  bf16*  xnB  = (bf16*)r2;
  bf16*  h1   = (bf16*)(r2 + SZ_BF_D);
  bf16*  h2   = (bf16*)(r2 + SZ_BF_D + SZ_BF_H);
  float* gate = (float*)r2;
  // region 3: x_dist f32 (phase 1)  ==aliased==  yT bf16 (phase 2)
  char* r3 = take(SZ_F32_H);
  float* xd = (float*)r3;
  bf16*  yT = (bf16*)r3;

  bf16* w1T = (bf16*)take((size_t)cH * cD * 2);
  bf16* w2T = (bf16*)take((size_t)cH * cH * 2);
  bf16* w3T = (bf16*)take((size_t)cDD * cH * 2);
  bf16* thT[2] = { (bf16*)take((size_t)cD*cD*2), (bf16*)take((size_t)cD*cD*2) };
  bf16* whT[2] = { (bf16*)take((size_t)cD*cD*2), (bf16*)take((size_t)cD*cD*2) };
  bf16* wtT[2] = { (bf16*)take((size_t)cD*cD*2), (bf16*)take((size_t)cD*cD*2) };

  int*   binIdx = (int*)  take((size_t)cBN * 4);
  int*   counts = (int*)  take((size_t)cB * cNKEY * 4);
  int*   offs   = (int*)  take((size_t)cB * cNKEY * 4);
  int*   split  = (int*)  take((size_t)cBN * 4);
  float* mskf   = (float*)take((size_t)cBN * 4);
  float* nsq    = (float*)take((size_t)cBN * 4);
  bf16*  xm     = (bf16*) take(SZ_BF_H);
  float* xbF    = (float*)take(SZ_F32_D);
  bf16*  xbB    = (bf16*) take(SZ_BF_D);
  bf16*  xbmB   = (bf16*) take(SZ_BF_D);
  bf16*  dmB    = (bf16*) take(SZ_BF_H);  // 1600 bins * 128*128 bf16

  // --- weights -> bf16 transposed ---
  k_transpose_bf16<<<(cD*cH + 255)/256, 256, 0, stream>>>(w1, w1T, cD, cH);
  k_transpose_bf16<<<(cH*cH + 255)/256, 256, 0, stream>>>(w2, w2T, cH, cH);
  k_transpose_bf16<<<(cH*cDD + 255)/256, 256, 0, stream>>>(w3, w3T, cH, cDD);
  for (int l = 0; l < 2; l++) {
    k_transpose_bf16<<<(cD*cD + 255)/256, 256, 0, stream>>>(thW[l], thT[l], cD, cD);
    k_transpose_bf16<<<(cD*cD + 255)/256, 256, 0, stream>>>(whW[l], whT[l], cD, cD);
    k_transpose_bf16<<<(cD*cD + 255)/256, 256, 0, stream>>>(wtW[l], wtT[l], cD, cD);
  }

  // --- layernorm ---
  k_layernorm<<<cBN, cD, 0, stream>>>(x, ln_g, ln_b, xnF, xnB);

  // --- distance FFN (WMMA, 32x32 macro-tiles per wave) ---
  const int gFFN = (cBN / 32) * (cH / 32) / 8;   // 3200 blocks
  k_gemm_ffn<<<gFFN, 256, 0, stream>>>(xnB, w1T, b1, h1, nullptr, cBN, cH, cD);
  k_gemm_ffn<<<gFFN, 256, 0, stream>>>(h1, w2T, b2, h2, nullptr, cBN, cH, cH);
  k_gemm_ffn<<<gFFN, 256, 0, stream>>>(h2, w3T, b3, nullptr, xd, cBN, cDD, cH);

  // --- LSH binning + deterministic stable counting sort ---
  k_lsh<<<cBN / 8, 256, 0, stream>>>(xd, cb, msk, binIdx);
  k_zero<<<(cB*cNKEY + 255)/256, 256, 0, stream>>>(counts, cB * cNKEY);
  k_hist<<<(cBN + 255)/256, 256, 0, stream>>>(binIdx, counts);
  k_scan<<<1, 32, 0, stream>>>(counts, offs);
  k_rank<<<cB * cNKEY, 32, 0, stream>>>(binIdx, offs, split);

  // --- gather into bins ---
  k_gather<<<cBN, 256, 0, stream>>>(split, msk, xnF, xd, mskf, nsq, xm,
                                    xbF, xbB, xbmB);

  // --- per-bin Gaussian kernel matrix (WMMA) ---
  k_dm<<<cB * cNB, 256, 0, stream>>>(xm, nsq, mskf, dmB);

  // --- two gated graph convolutions (WMMA) ---
  const int gBIG = (cBN / 32) * (cD / 32) / 8;   // 6400 blocks
  for (int l = 0; l < 2; l++) {
    k_gemm_big<<<gBIG, 256, 0, stream>>>(xbmB, thT[l], nullptr, mskf, yT, nullptr, 0);
    k_gemm_big<<<gBIG, 256, 0, stream>>>(xbmB, whT[l], nullptr, nullptr, nullptr, fhet, 1);
    k_gemm_big<<<gBIG, 256, 0, stream>>>(xbB,  wtT[l], btW[l],  nullptr, nullptr, gate, 2);
    k_hom<<<cB * cNB, 256, 0, stream>>>(dmB, yT, gate, fhet, mskf, xbF, xbB, xbmB);
  }

  // --- reverse scatter ---
  k_scatter<<<cBN, cD, 0, stream>>>(split, xbF, out);
}
